// PostProcessor_73804718014899
// MI455X (gfx1250) — compile-verified, matrix operand order checked by reading, not run on hardware
//
#include <hip/hip_runtime.h>
#include <hip/hip_bf16.h>
#include <stdint.h>

// SSD post-processor for MI455X (gfx1250, wave32).
// d_in[0] = cls_scores [8,8732,21] f32
// d_in[1] = bbox_pred  [8,8732,4]  f32
// d_in[2] = anchors    [8732,4]    f32
// d_out   = boxes[8,200,4] ++ scores[8,200] ++ labels[8,200]  (f32, 9600 elems)

#define NA     8732
#define NC     21
#define FG     20
#define BATCH  8
#define PRE    1024
#define TOPK   200
#define SORTN  16384
#define GSORTN 4096
#define CONF   0.5f
#define NMS_T  0.45f
#define IMGSZ  300.0f
// coords scaled by 1/16 so f16 areas (<= 90601/256) stay in range for WMMA
#define CSCALE 0.0625f
#define ONE_S  0.0625f
#define NEG_INF (-__builtin_inff())

typedef __attribute__((ext_vector_type(16))) _Float16 v16h;
typedef __attribute__((ext_vector_type(8)))  float    v8f;
typedef __attribute__((ext_vector_type(4)))  uint32_t u32x4;
typedef __attribute__((ext_vector_type(8)))  int      i32x8;
typedef __attribute__((ext_vector_type(4)))  int      i32x4;

// ---------------------------------------------------------------- decode
__global__ __launch_bounds__(256) void ssd_decode_kernel(
    const float* __restrict__ loc, const float* __restrict__ anchors,
    float* __restrict__ dec, int total) {
  int idx = blockIdx.x * blockDim.x + threadIdx.x;
  if (idx >= total) return;
  int a = idx % NA;
  __builtin_prefetch(loc + (size_t)idx * 4 + 256, 0, 3);  // global_prefetch_b8
  float lx = loc[idx * 4 + 0], ly = loc[idx * 4 + 1];
  float lw = loc[idx * 4 + 2], lh = loc[idx * 4 + 3];
  float acx = anchors[a * 4 + 0], acy = anchors[a * 4 + 1];
  float aw = anchors[a * 4 + 2], ah = anchors[a * 4 + 3];
  float cx = acx + lx * 0.1f * aw;
  float cy = acy + ly * 0.1f * ah;
  float w  = aw * expf(lw * 0.2f);
  float h  = ah * expf(lh * 0.2f);
  float x1 = (cx - 0.5f * w) * IMGSZ;
  float y1 = (cy - 0.5f * h) * IMGSZ;
  float x2 = x1 + w * IMGSZ;
  float y2 = y1 + h * IMGSZ;
  dec[idx * 4 + 0] = fminf(fmaxf(x1, 0.f), IMGSZ - 1.f);
  dec[idx * 4 + 1] = fminf(fmaxf(y1, 0.f), IMGSZ - 1.f);
  dec[idx * 4 + 2] = fminf(fmaxf(x2, 0.f), IMGSZ - 1.f);
  dec[idx * 4 + 3] = fminf(fmaxf(y2, 0.f), IMGSZ - 1.f);
}

// ------------------------------------------------- per-(image,class) top-1024
__global__ __launch_bounds__(256) void ssd_topk_kernel(
    const float* __restrict__ cls, const float* __restrict__ dec,
    float* __restrict__ cand_boxes, float* __restrict__ cand_scores) {
  __shared__ float    skey[SORTN];   // 64 KB
  __shared__ unsigned sidx[SORTN];   // 64 KB
  const int bc = blockIdx.x;
  const int b = bc / FG, c = bc % FG;   // foreground class = c+1
  const int tid = threadIdx.x;

  for (int i = tid; i < SORTN; i += 256) {
    float v = NEG_INF; unsigned id = 0xFFFFFFFFu;
    if (i < NA) {
      float s = cls[((size_t)b * NA + i) * NC + (c + 1)];
      v = (s > CONF) ? s : NEG_INF;
      id = (unsigned)i;
    }
    skey[i] = v; sidx[i] = id;
  }
  __syncthreads();

  for (int k = 2; k <= SORTN; k <<= 1) {
    for (int j = k >> 1; j > 0; j >>= 1) {
      for (int i = tid; i < SORTN; i += 256) {
        int ixj = i ^ j;
        if (ixj > i) {
          bool desc = ((i & k) == 0);
          float a = skey[i], bb = skey[ixj];
          if (desc ? (a < bb) : (a > bb)) {
            skey[i] = bb; skey[ixj] = a;
            unsigned t = sidx[i]; sidx[i] = sidx[ixj]; sidx[ixj] = t;
          }
        }
      }
      __syncthreads();
    }
  }

  for (int kk = tid; kk < PRE; kk += 256) {
    float s = skey[kk]; unsigned id = sidx[kk];
    cand_scores[(size_t)bc * PRE + kk] = s;
    float bx0 = 0.f, bx1 = 0.f, bx2 = 0.f, bx3 = 0.f;
    if (id < NA) {
      bx0 = dec[((size_t)b * NA + id) * 4 + 0];
      bx1 = dec[((size_t)b * NA + id) * 4 + 1];
      bx2 = dec[((size_t)b * NA + id) * 4 + 2];
      bx3 = dec[((size_t)b * NA + id) * 4 + 3];
    }
    cand_boxes[((size_t)bc * PRE + kk) * 4 + 0] = bx0;
    cand_boxes[((size_t)bc * PRE + kk) * 4 + 1] = bx1;
    cand_boxes[((size_t)bc * PRE + kk) * 4 + 2] = bx2;
    cand_boxes[((size_t)bc * PRE + kk) * 4 + 3] = bx3;
  }
}

// ------------------------- NMS: TDM-staged boxes, WMMA area-sums, LDS bitmask
__global__ __launch_bounds__(256) void ssd_nms_kernel(
    const float* __restrict__ cand_boxes, const float* __restrict__ cand_scores,
    float* __restrict__ comp_boxes, float* __restrict__ comp_scores) {
  __shared__ float    sbox[PRE * 4];      // 16 KB  (coords * 1/16)
  __shared__ float    sscore[PRE];        // 4 KB
  __shared__ unsigned smask[PRE * 32];    // 128 KB suppression bit-matrix
  __shared__ unsigned skept[PRE];         // 4 KB

  const int bc = blockIdx.x;
  const int tid = threadIdx.x;
  const int lane = tid & 31;
  const int wave = tid >> 5;

  for (int i = tid; i < PRE * 32; i += 256) smask[i] = 0u;
  for (int i = tid; i < PRE; i += 256)
    sscore[i] = cand_scores[(size_t)bc * PRE + i];

  // Tensor Data Mover: DMA the 16 KB candidate-box tile global -> LDS.
  if (wave == 0) {
    unsigned lds_addr = (unsigned)(uintptr_t)(&sbox[0]);
    unsigned long long ga =
        (unsigned long long)(uintptr_t)(cand_boxes + (size_t)bc * PRE * 4);
    u32x4 g0 = {1u,                                    // count=1 (valid D#)
                lds_addr,                              // LDS dest (bytes)
                (uint32_t)(ga & 0xFFFFFFFFull),        // global_addr lo
                (uint32_t)((ga >> 32) & 0x01FFFFFFull) // global_addr hi
                    | 0x80000000u};                    // type=2 ("image")
    // 1 row x 4096 f32: data_size=4B, tensor_dim0=4096, tile_dim0=4096,
    // tensor_dim1=1, tile_dim1=1, strides=4096.
    i32x8 g1 = {0x00020000,   // data_size=2 (4 bytes)
                0x10000000,   // tensor_dim0[15:0]=4096 in bits[31:16]
                0x00010000,   // tensor_dim0 hi=0; tensor_dim1[15:0]=1
                0x10000000,   // tensor_dim1 hi=0; tile_dim0=4096
                0x00000001,   // tile_dim1=1; tile_dim2=0
                4096,         // tensor_dim0_stride[31:0]
                0x10000000,   // stride0 hi=0; tensor_dim1_stride lo=4096
                0};
    i32x4 gz = {0, 0, 0, 0};
#if __clang_major__ >= 23
    i32x8 gz8 = {0, 0, 0, 0, 0, 0, 0, 0};
    __builtin_amdgcn_tensor_load_to_lds(g0, g1, gz, gz, gz8, 0);
#else
    __builtin_amdgcn_tensor_load_to_lds(g0, g1, gz, gz, 0);
#endif
    __builtin_amdgcn_s_wait_tensorcnt(0);
  }
  __syncthreads();
  for (int i = tid; i < PRE * 4; i += 256) sbox[i] *= CSCALE;
  __syncthreads();

  // 64x64 grid of 16x16 tiles over the IoU matrix; upper-triangle tiles only.
  // WMMA computes the pairwise area-sum (IoU denominator core) per tile:
  //   D = area_row * ones^T + ones * area_col^T   (two chained v_wmma).
  int t = 0;
  for (int ti = 0; ti < PRE / 16; ++ti) {
    for (int tj = ti; tj < PRE / 16; ++tj, ++t) {
      if ((t & 7) != wave) continue;              // uniform per wave
      const int rowbase = ti * 16, colbase = tj * 16;
      v16h a1 = {}, b1 = {}, a2 = {}, b2 = {};
      if (lane < 16) {
        int r = rowbase + lane;
        float ar = (sbox[r * 4 + 2] - sbox[r * 4 + 0] + ONE_S) *
                   (sbox[r * 4 + 3] - sbox[r * 4 + 1] + ONE_S);
        int c = colbase + lane;
        float ac = (sbox[c * 4 + 2] - sbox[c * 4 + 0] + ONE_S) *
                   (sbox[c * 4 + 3] - sbox[c * 4 + 1] + ONE_S);
        a1[0] = (_Float16)ar;    // A: row areas in K=0 column
        b1[0] = (_Float16)1.0f;  // B: ones in K=0 row
        a2[0] = (_Float16)1.0f;  // A: ones in K=0 column
        b2[0] = (_Float16)ac;    // B: col areas in K=0 row
      }
      v8f acc = {};
      acc = __builtin_amdgcn_wmma_f32_16x16x32_f16(false, a1, false, b1,
                                                   (short)0, acc, false, false);
      acc = __builtin_amdgcn_wmma_f32_16x16x32_f16(false, a2, false, b2,
                                                   (short)0, acc, false, false);
      // C/D layout: acc[v] at (i = rowbase + v + 8*(lane/16), j = colbase + lane%16)
      const int j = colbase + (lane & 15);
      const float jx1 = sbox[j * 4 + 0], jy1 = sbox[j * 4 + 1];
      const float jx2 = sbox[j * 4 + 2], jy2 = sbox[j * 4 + 3];
      const int ibase = rowbase + ((lane >> 4) << 3);
#pragma unroll
      for (int v = 0; v < 8; ++v) {
        int i = ibase + v;
        if (j > i) {
          float w = fminf(sbox[i * 4 + 2], jx2) - fmaxf(sbox[i * 4 + 0], jx1) + ONE_S;
          float h = fminf(sbox[i * 4 + 3], jy2) - fmaxf(sbox[i * 4 + 1], jy1) + ONE_S;
          w = fmaxf(w, 0.f); h = fmaxf(h, 0.f);
          float inter = w * h;
          float iou = inter / (acc[v] - inter);
          if (iou > NMS_T)
            atomicOr(&smask[i * 32 + (j >> 5)], 1u << (j & 31));
        }
      }
    }
  }
  __syncthreads();

  // Greedy reduction: one wave32; lane l owns 32-column block l.
  if (wave == 0) {
    unsigned remv = 0u;
    for (int i = 0; i < PRE; ++i) {
      int blk = __builtin_amdgcn_ds_bpermute((i >> 5) << 2, (int)remv);
      bool sup = (((unsigned)blk >> (i & 31)) & 1u) != 0u;
      bool keep = (!sup) && (sscore[i] > CONF);   // uniform across wave
      if (keep) remv |= smask[i * 32 + lane];
      if (lane == 0) skept[i] = keep ? 1u : 0u;
    }
  }
  __syncthreads();

  // Compact first 200 survivors (already score-descending), pad with -inf.
  if (tid == 0) {
    int k = 0;
    const size_t ob = (size_t)bc * TOPK;
    for (int i = 0; i < PRE && k < TOPK; ++i) {
      if (skept[i]) {
        comp_scores[ob + k] = sscore[i];
        comp_boxes[(ob + k) * 4 + 0] = sbox[i * 4 + 0] * 16.0f;
        comp_boxes[(ob + k) * 4 + 1] = sbox[i * 4 + 1] * 16.0f;
        comp_boxes[(ob + k) * 4 + 2] = sbox[i * 4 + 2] * 16.0f;
        comp_boxes[(ob + k) * 4 + 3] = sbox[i * 4 + 3] * 16.0f;
        ++k;
      }
    }
    for (; k < TOPK; ++k) {
      comp_scores[ob + k] = NEG_INF;
      comp_boxes[(ob + k) * 4 + 0] = 0.f;
      comp_boxes[(ob + k) * 4 + 1] = 0.f;
      comp_boxes[(ob + k) * 4 + 2] = 0.f;
      comp_boxes[(ob + k) * 4 + 3] = 0.f;
    }
  }
}

// ------------------------------------------- per-image global top-200 + output
__global__ __launch_bounds__(256) void ssd_final_kernel(
    const float* __restrict__ comp_boxes, const float* __restrict__ comp_scores,
    float* __restrict__ out) {
  __shared__ float    gkey[GSORTN];   // 16 KB
  __shared__ unsigned gval[GSORTN];   // 16 KB
  const int b = blockIdx.x, tid = threadIdx.x;

  for (int i = tid; i < GSORTN; i += 256) {
    float v = NEG_INF; unsigned id = 0u;
    if (i < FG * TOPK) { v = comp_scores[(size_t)b * FG * TOPK + i]; id = (unsigned)i; }
    gkey[i] = v; gval[i] = id;
  }
  __syncthreads();

  for (int k = 2; k <= GSORTN; k <<= 1) {
    for (int j = k >> 1; j > 0; j >>= 1) {
      for (int i = tid; i < GSORTN; i += 256) {
        int ixj = i ^ j;
        if (ixj > i) {
          bool desc = ((i & k) == 0);
          float a = gkey[i], c = gkey[ixj];
          if (desc ? (a < c) : (a > c)) {
            gkey[i] = c; gkey[ixj] = a;
            unsigned tv = gval[i]; gval[i] = gval[ixj]; gval[ixj] = tv;
          }
        }
      }
      __syncthreads();
    }
  }

  float* out_boxes  = out;
  float* out_scores = out + (size_t)BATCH * TOPK * 4;
  float* out_labels = out + (size_t)BATCH * TOPK * 4 + (size_t)BATCH * TOPK;
  for (int kk = tid; kk < TOPK; kk += 256) {
    float s = gkey[kk];
    bool valid = s > NEG_INF;
    unsigned id = gval[kk];
    float bx[4] = {0.f, 0.f, 0.f, 0.f};
    int label = 0;
    if (valid) {
      label = (int)(id / TOPK) + 1;
#pragma unroll
      for (int d = 0; d < 4; ++d)
        bx[d] = comp_boxes[((size_t)b * FG * TOPK + id) * 4 + d];
    }
#pragma unroll
    for (int d = 0; d < 4; ++d)
      out_boxes[((size_t)b * TOPK + kk) * 4 + d] = bx[d];
    out_scores[(size_t)b * TOPK + kk] = valid ? s : 0.f;
    out_labels[(size_t)b * TOPK + kk] = (float)label;
  }
}

// ---------------------------------------------------------------- launcher
extern "C" void kernel_launch(void* const* d_in, const int* in_sizes, int n_in,
                              void* d_out, int out_size, void* d_ws, size_t ws_size,
                              hipStream_t stream) {
  (void)in_sizes; (void)n_in; (void)out_size; (void)ws_size;
  const float* cls     = (const float*)d_in[0];
  const float* loc     = (const float*)d_in[1];
  const float* anchors = (const float*)d_in[2];

  // workspace layout (floats): ~4.8 MB total
  float* dec         = (float*)d_ws;                         // B*NA*4
  float* cand_boxes  = dec + (size_t)BATCH * NA * 4;         // B*FG*PRE*4
  float* cand_scores = cand_boxes + (size_t)BATCH * FG * PRE * 4;  // B*FG*PRE
  float* comp_boxes  = cand_scores + (size_t)BATCH * FG * PRE;     // B*FG*TOPK*4
  float* comp_scores = comp_boxes + (size_t)BATCH * FG * TOPK * 4; // B*FG*TOPK

  int total = BATCH * NA;
  ssd_decode_kernel<<<(total + 255) / 256, 256, 0, stream>>>(loc, anchors, dec, total);
  ssd_topk_kernel<<<BATCH * FG, 256, 0, stream>>>(cls, dec, cand_boxes, cand_scores);
  ssd_nms_kernel<<<BATCH * FG, 256, 0, stream>>>(cand_boxes, cand_scores,
                                                 comp_boxes, comp_scores);
  ssd_final_kernel<<<BATCH, 256, 0, stream>>>(comp_boxes, comp_scores, (float*)d_out);
}
